// GCNReg_33406255628686
// MI455X (gfx1250) — matrix operand order repaired and checked by reading, not compiled, not run on hardware
//
#include <hip/hip_runtime.h>

#define N_NODES 50000
#define N_EDGES 800000
#define CH      512
#define MTILES  (N_NODES / 16)   // 3125

typedef __attribute__((ext_vector_type(16))) __bf16 v16bf;
typedef __attribute__((ext_vector_type(8)))  float  v8f;

// ---------------- degree / normalization ----------------
__global__ void k_zero_deg(float* deg) {
  int i = blockIdx.x * blockDim.x + threadIdx.x;
  if (i < N_NODES) deg[i] = 0.0f;
}

__global__ void k_degree(const long long* __restrict__ ei, float* __restrict__ deg) {
  int e = blockIdx.x * blockDim.x + threadIdx.x;
  if (e < N_EDGES) {
    int col = (int)ei[N_EDGES + e];
    atomicAdd(deg + col, 1.0f);
  }
}

__global__ void k_dis(float* degdis) {
  int i = blockIdx.x * blockDim.x + threadIdx.x;
  if (i < N_NODES) degdis[i] = rsqrtf(degdis[i] + 1.0f);  // +1 self loop
}

// ---------------- pack W1 -> bf16 B-fragments ----------------
// Fragment (kstep in [0,16), ntile in [0,32)): per lane (wave32):
//   col = ntile*16 + (lane&15), kb = kstep*32 + (lane>>4)*8
//   element j = W1[kb+j+(j>=8?8:0)][col]   (B is 32x16, K-major per ISA layout)
__global__ void k_pack_w1(const float* __restrict__ W1, __bf16* __restrict__ packedB) {
  int wave = blockIdx.x * 8 + (threadIdx.x >> 5);   // 64 blocks * 8 waves = 512 frags
  int lane = threadIdx.x & 31;
  int kstep = wave >> 5;
  int ntile = wave & 31;
  int n  = ntile * 16 + (lane & 15);
  int kb = kstep * 32 + ((lane >> 4) << 3);
  v16bf frag;
#pragma unroll
  for (int j = 0; j < 16; ++j)
    frag[j] = (__bf16)W1[(size_t)(kb + j + (j >= 8 ? 8 : 0)) * CH + n];
  reinterpret_cast<v16bf*>(packedB)[wave * 32 + lane] = frag;
}

// ---------------- GEMM1: xw = bf16(x) @ bf16(W1), fp32 accum ----------------
// One wave -> 16(M) x 64(N) strip; grid = (ceil(3125/8), 8 y-strips of 64 cols)
__global__ void __launch_bounds__(256) k_gemm1(const float* __restrict__ x,
                                               const __bf16* __restrict__ packedB,
                                               float* __restrict__ xw) {
  const int lane  = threadIdx.x & 31;
  const int wv    = threadIdx.x >> 5;
  const int mtile = blockIdx.x * 8 + wv;
  if (mtile >= MTILES) return;
  const int m0 = mtile * 16;
  const int nb = blockIdx.y * 64;
  const int arow = m0 + (lane & 15);
  const int ka   = (lane >> 4) * 8;          // K split per ISA A layout
  const v16bf* B = reinterpret_cast<const v16bf*>(packedB);

  v8f acc0 = {}, acc1 = {}, acc2 = {}, acc3 = {};
#pragma unroll 4
  for (int kstep = 0; kstep < CH / 32; ++kstep) {
    const float* ap = x + (size_t)arow * CH + kstep * 32 + ka;
    v16bf a;
#pragma unroll
    for (int j = 0; j < 8; ++j) {
      a[j]     = (__bf16)ap[j];        // K = ka+0..7
      a[j + 8] = (__bf16)ap[j + 16];   // K = ka+16..23
    }
    const int fbase = (kstep * 32 + (nb >> 4)) * 32 + lane;
    v16bf b0 = B[fbase];
    v16bf b1 = B[fbase + 32];
    v16bf b2 = B[fbase + 64];
    v16bf b3 = B[fbase + 96];
    acc0 = __builtin_amdgcn_wmma_f32_16x16x32_bf16(false, a, false, b0, (short)0, acc0, false, false);
    acc1 = __builtin_amdgcn_wmma_f32_16x16x32_bf16(false, a, false, b1, (short)0, acc1, false, false);
    acc2 = __builtin_amdgcn_wmma_f32_16x16x32_bf16(false, a, false, b2, (short)0, acc2, false, false);
    acc3 = __builtin_amdgcn_wmma_f32_16x16x32_bf16(false, a, false, b3, (short)0, acc3, false, false);
  }
  const int drow0 = m0 + ((lane >> 4) * 8);
  const int dcol  = nb + (lane & 15);
#pragma unroll
  for (int i = 0; i < 8; ++i) {
    float* dst = xw + (size_t)(drow0 + i) * CH + dcol;
    dst[0]  = acc0[i];
    dst[16] = acc1[i];
    dst[32] = acc2[i];
    dst[48] = acc3[i];
  }
}

// ---------------- agg init with self-loop: agg = dis^2 * xw ----------------
__global__ void k_selfinit(const float* __restrict__ xw, const float* __restrict__ dis,
                           float* __restrict__ agg) {
  int idx = blockIdx.x * blockDim.x + threadIdx.x;      // float4 index
  if (idx >= N_NODES * (CH / 4)) return;
  int node = idx / (CH / 4);
  float d = dis[node];
  float s = d * d;
  float4 v = reinterpret_cast<const float4*>(xw)[idx];
  float4 r = make_float4(v.x * s, v.y * s, v.z * s, v.w * s);
  reinterpret_cast<float4*>(agg)[idx] = r;
}

// ---------------- edge scatter layer1: one wave per edge ----------------
__global__ void k_scatter1(const long long* __restrict__ ei, const float* __restrict__ dis,
                           const float* __restrict__ xw, float* __restrict__ agg) {
  int e    = (blockIdx.x * blockDim.x + threadIdx.x) >> 5;
  int lane = threadIdx.x & 31;
  if (e >= N_EDGES) return;
  int row = (int)ei[e];
  int col = (int)ei[N_EDGES + e];
  float norm = dis[row] * dis[col];
  const float4* src = reinterpret_cast<const float4*>(xw + (size_t)row * CH);
  float* dst = agg + (size_t)col * CH;
#pragma unroll
  for (int j = 0; j < 4; ++j) {
    int i4 = j * 32 + lane;                 // coalesced 512B per iteration
    float4 v = src[i4];
    atomicAdd(dst + i4 * 4 + 0, norm * v.x);
    atomicAdd(dst + i4 * 4 + 1, norm * v.y);
    atomicAdd(dst + i4 * 4 + 2, norm * v.z);
    atomicAdd(dst + i4 * 4 + 3, norm * v.w);
  }
}

// ---------------- h = relu(agg + b1) in place ----------------
__global__ void k_relu_bias(float* __restrict__ agg, const float* __restrict__ b1) {
  int idx = blockIdx.x * blockDim.x + threadIdx.x;      // float4 index
  if (idx >= N_NODES * (CH / 4)) return;
  int c4 = idx & (CH / 4 - 1);
  float4 b = reinterpret_cast<const float4*>(b1)[c4];
  float4 v = reinterpret_cast<float4*>(agg)[idx];
  v.x = fmaxf(v.x + b.x, 0.0f);
  v.y = fmaxf(v.y + b.y, 0.0f);
  v.z = fmaxf(v.z + b.z, 0.0f);
  v.w = fmaxf(v.w + b.w, 0.0f);
  reinterpret_cast<float4*>(agg)[idx] = v;
}

// ---------------- hw2 = h @ W2 (wave-per-node dot over 512) ----------------
__global__ void k_gemv2(const float* __restrict__ h, const float* __restrict__ W2,
                        float* __restrict__ hw2) {
  int node = (blockIdx.x * blockDim.x + threadIdx.x) >> 5;
  int lane = threadIdx.x & 31;
  if (node >= N_NODES) return;
  const float4* hp = reinterpret_cast<const float4*>(h + (size_t)node * CH);
  const float4* wp = reinterpret_cast<const float4*>(W2);
  float s = 0.0f;
#pragma unroll
  for (int j = 0; j < 4; ++j) {
    int i4 = j * 32 + lane;
    float4 a = hp[i4];
    float4 b = wp[i4];
    s += a.x * b.x + a.y * b.y + a.z * b.z + a.w * b.w;
  }
#pragma unroll
  for (int off = 16; off > 0; off >>= 1) s += __shfl_xor(s, off, 32);
  if (lane == 0) hw2[node] = s;
}

// ---------------- out init (self-loop + bias) and layer-2 scatter ----------
__global__ void k_out_init(const float* __restrict__ dis, const float* __restrict__ hw2,
                           const float* __restrict__ b2, float* __restrict__ out) {
  int i = blockIdx.x * blockDim.x + threadIdx.x;
  if (i < N_NODES) {
    float d = dis[i];
    out[i] = d * d * hw2[i] + b2[0];
  }
}

__global__ void k_scatter2(const long long* __restrict__ ei, const float* __restrict__ dis,
                           const float* __restrict__ hw2, float* __restrict__ out) {
  int e = blockIdx.x * blockDim.x + threadIdx.x;
  if (e < N_EDGES) {
    int row = (int)ei[e];
    int col = (int)ei[N_EDGES + e];
    atomicAdd(out + col, dis[row] * dis[col] * hw2[row]);
  }
}

extern "C" void kernel_launch(void* const* d_in, const int* in_sizes, int n_in,
                              void* d_out, int out_size, void* d_ws, size_t ws_size,
                              hipStream_t stream) {
  const float*     x  = (const float*)d_in[0];
  const long long* ei = (const long long*)d_in[1];   // int64 [2, E]
  const float*     W1 = (const float*)d_in[2];
  const float*     b1 = (const float*)d_in[3];
  const float*     W2 = (const float*)d_in[4];
  const float*     b2 = (const float*)d_in[5];
  float* out = (float*)d_out;

  // workspace layout (~206 MB):
  char*   ws      = (char*)d_ws;
  __bf16* packedB = (__bf16*)ws;                                // 512*512*2 = 524288 B
  float*  xw      = (float*)(ws + 524288);                      // N*CH f32
  float*  agg     = xw  + (size_t)N_NODES * CH;                 // N*CH f32 (becomes h)
  float*  dis     = agg + (size_t)N_NODES * CH;                 // N f32 (deg -> dis)
  float*  hw2     = dis + N_NODES;                              // N f32

  k_zero_deg<<<(N_NODES + 255) / 256, 256, 0, stream>>>(dis);
  k_pack_w1 <<<64, 256, 0, stream>>>(W1, packedB);
  k_degree  <<<(N_EDGES + 255) / 256, 256, 0, stream>>>(ei, dis);
  k_dis     <<<(N_NODES + 255) / 256, 256, 0, stream>>>(dis);

  dim3 ggrid((MTILES + 7) / 8, CH / 64);
  k_gemm1   <<<ggrid, 256, 0, stream>>>(x, packedB, xw);

  k_selfinit<<<(N_NODES * (CH / 4) + 255) / 256, 256, 0, stream>>>(xw, dis, agg);
  k_scatter1<<<(N_EDGES * 32) / 256, 256, 0, stream>>>(ei, dis, xw, agg);
  k_relu_bias<<<(N_NODES * (CH / 4) + 255) / 256, 256, 0, stream>>>(agg, b1);
  k_gemv2   <<<(N_NODES * 32 + 255) / 256, 256, 0, stream>>>(agg, W2, hw2);
  k_out_init<<<(N_NODES + 255) / 256, 256, 0, stream>>>(dis, hw2, b2, out);
  k_scatter2<<<(N_EDGES + 255) / 256, 256, 0, stream>>>(ei, dis, hw2, out);
}
